// EnhancedAdaptiveRecursiveCell_24902220382621
// MI455X (gfx1250) — matrix operand order repaired and checked by reading, not compiled, not run on hardware
//
#include <hip/hip_runtime.h>
#include <hip/hip_bf16.h>

#define DIMN  2048
#define BATCH 16384
#define EPSF  1e-6f

typedef __bf16 bf16_t;
typedef __attribute__((ext_vector_type(16))) __bf16 v16bf;
typedef __attribute__((ext_vector_type(8)))  float  v8f;

#define TM 64
#define TN 128
#define TK 32
#define AP 40   // padded LDS K-stride (bf16): 80B rows, 16B-aligned, conflict-free b128

union Frag16 { v16bf v; uint4 q[2]; };

// generic LDS pointer -> 32-bit LDS offset (aperture low bits are the DS address)
__device__ __forceinline__ unsigned ldsOff(const void* p) {
  return (unsigned)(unsigned long long)p;
}

// async global -> LDS copy, 16B per lane; IOFFSET applies to BOTH addresses (ISA 8.2)
template<int OFF>
__device__ __forceinline__ void async_b128(unsigned lds, const void* g) {
  if constexpr (OFF == 0)
    asm volatile("global_load_async_to_lds_b128 %0, %1, off"
                 :: "v"(lds), "v"(g) : "memory");
  else if constexpr (OFF == 16)
    asm volatile("global_load_async_to_lds_b128 %0, %1, off offset:16"
                 :: "v"(lds), "v"(g) : "memory");
  else if constexpr (OFF == 32)
    asm volatile("global_load_async_to_lds_b128 %0, %1, off offset:32"
                 :: "v"(lds), "v"(g) : "memory");
  else
    asm volatile("global_load_async_to_lds_b128 %0, %1, off offset:48"
                 :: "v"(lds), "v"(g) : "memory");
}

template<int N>
__device__ __forceinline__ void wait_async_le() {
  if constexpr (N == 0)      asm volatile("s_wait_asynccnt 0x0" ::: "memory");
  else if constexpr (N == 4) asm volatile("s_wait_asynccnt 0x4" ::: "memory");
  else                       asm volatile("s_wait_asynccnt 0x6" ::: "memory");
}

// ---------------------------------------------------------------------------
// Complex GEMM:  Cr + i*Ci = (Ar + i*Ai) @ (Wr + i*Wi)
// B matrices pre-transposed bf16 [N][K], ld = DIMN.
// Block tile 64x128, 8 waves of (16x64), K-step 32, double-buffered LDS,
// async global->LDS copies for bf16 tiles, v_wmma_f32_16x16x32_bf16.
// ---------------------------------------------------------------------------
template<bool A_F32, bool OUT_F32>
__global__ __launch_bounds__(256) void cgemm_kernel(
    const void* __restrict__ Ar_, const void* __restrict__ Ai_, long ldaE,
    const void* __restrict__ Br_, const void* __restrict__ Bi_,
    void* __restrict__ Cr_, void* __restrict__ Ci_, long ldcE)
{
  __shared__ __align__(16) bf16_t sA[2][2][TM * AP];  // [buf][re/im]
  __shared__ __align__(16) bf16_t sB[2][2][TN * AP];

  const int t    = threadIdx.x;
  const int lane = t & 31;
  const int wave = t >> 5;
  const int wm   = wave >> 1;          // 0..3 : 16-row slab
  const int wn   = wave & 1;           // 0..1 : 64-col slab
  const int half = lane >> 4;
  const int mlow = lane & 15;

  const long m0 = (long)blockIdx.y * TM;
  const long n0 = (long)blockIdx.x * TN;

  const bf16_t* Br = (const bf16_t*)Br_;
  const bf16_t* Bi = (const bf16_t*)Bi_;

  // acc banks: zr = RP - RM ; zi = IP + IQ  (bf16 WMMA has no A/B negate)
  v8f accRP[4] = {};
  v8f accRM[4] = {};
  v8f accIP[4] = {};
  v8f accIQ[4] = {};

  // cooperative-fill roles
  const int matSel = t >> 7;       // 0 real / 1 imag
  const int idx    = t & 127;
  const int lrow   = idx >> 1;     // A: row 0..63
  const int lhalf  = idx & 1;      // A: 16-elem half of the 32-K slab
  const int bn     = idx;          // B: n 0..127

  // issues 6 async copies/wave per tile (bf16 A) or 4 (fp32 A uses ds path)
  auto fillTile = [&](int buf, int kk) {
    // ---- A tile (64 x 32 bf16) ----
    if constexpr (A_F32) {
      const float* src = (const float*)(matSel ? Ai_ : Ar_) +
                         (m0 + lrow) * ldaE + kk + lhalf * 16;
      float4 f0 = ((const float4*)src)[0];
      float4 f1 = ((const float4*)src)[1];
      float4 f2 = ((const float4*)src)[2];
      float4 f3 = ((const float4*)src)[3];
      __align__(16) bf16_t tmp[16];
      tmp[0]=(bf16_t)f0.x; tmp[1]=(bf16_t)f0.y; tmp[2]=(bf16_t)f0.z; tmp[3]=(bf16_t)f0.w;
      tmp[4]=(bf16_t)f1.x; tmp[5]=(bf16_t)f1.y; tmp[6]=(bf16_t)f1.z; tmp[7]=(bf16_t)f1.w;
      tmp[8]=(bf16_t)f2.x; tmp[9]=(bf16_t)f2.y; tmp[10]=(bf16_t)f2.z; tmp[11]=(bf16_t)f2.w;
      tmp[12]=(bf16_t)f3.x; tmp[13]=(bf16_t)f3.y; tmp[14]=(bf16_t)f3.z; tmp[15]=(bf16_t)f3.w;
      bf16_t* dst = &sA[buf][matSel][lrow * AP + lhalf * 16];
      ((uint4*)dst)[0] = *(const uint4*)(tmp);
      ((uint4*)dst)[1] = *(const uint4*)(tmp + 8);
      if (kk + 2 * TK < DIMN) __builtin_prefetch(src + 2 * TK, 0, 1);
    } else {
      const bf16_t* src = (const bf16_t*)(matSel ? Ai_ : Ar_) +
                          (m0 + lrow) * ldaE + kk + lhalf * 16;
      const unsigned dst = ldsOff(&sA[buf][matSel][lrow * AP + lhalf * 16]);
      async_b128<0>(dst, src);
      async_b128<16>(dst, src);
    }
    // ---- B tile (128 x 32 bf16, n-major): 64B per thread, async ----
    {
      const bf16_t* src = (matSel ? Bi : Br) + (n0 + bn) * (long)DIMN + kk;
      const unsigned dst = ldsOff(&sB[buf][matSel][bn * AP]);
      async_b128<0>(dst, src);
      async_b128<16>(dst, src);
      async_b128<32>(dst, src);
      async_b128<48>(dst, src);
    }
  };

  constexpr int PER_TILE = A_F32 ? 4 : 6;
  constexpr int nIter = DIMN / TK;

  fillTile(0, 0);

  for (int it = 0; it < nIter; ++it) {
    const int cur = it & 1;
    if (it + 1 < nIter) {
      fillTile(cur ^ 1, (it + 1) * TK);     // prefetch next tile into spare buffer
      wait_async_le<PER_TILE>();            // only the new copies may be in flight
    } else {
      wait_async_le<0>();
    }
    __syncthreads();

    // A frag (16x32): lane<16 -> row=mlow, K 0..7 & 16..23 ; lane>=16 -> K 8..15 & 24..31
    const bf16_t* arow = &sA[cur][0][(wm * 16 + mlow) * AP + half * 8];
    const bf16_t* irow = &sA[cur][1][(wm * 16 + mlow) * AP + half * 8];
    Frag16 far, fai;
    far.q[0] = *(const uint4*)(arow);
    far.q[1] = *(const uint4*)(arow + 16);
    fai.q[0] = *(const uint4*)(irow);
    fai.q[1] = *(const uint4*)(irow + 16);

#pragma unroll
    for (int s = 0; s < 4; ++s) {
      // B frag (32x16): lane = column n, lane<16 -> K 0..7 & 16..23 ; lane>=16 -> K 8..15 & 24..31
      const bf16_t* brow = &sB[cur][0][(wn * 64 + s * 16 + mlow) * AP + half * 8];
      const bf16_t* bior = &sB[cur][1][(wn * 64 + s * 16 + mlow) * AP + half * 8];
      Frag16 fbr, fbi;
      fbr.q[0] = *(const uint4*)(brow);
      fbr.q[1] = *(const uint4*)(brow + 16);
      fbi.q[0] = *(const uint4*)(bior);
      fbi.q[1] = *(const uint4*)(bior + 16);

      accRP[s] = __builtin_amdgcn_wmma_f32_16x16x32_bf16(false, far.v, false, fbr.v,
                                                         (short)0, accRP[s], false, false);
      accRM[s] = __builtin_amdgcn_wmma_f32_16x16x32_bf16(false, fai.v, false, fbi.v,
                                                         (short)0, accRM[s], false, false);
      accIP[s] = __builtin_amdgcn_wmma_f32_16x16x32_bf16(false, fai.v, false, fbr.v,
                                                         (short)0, accIP[s], false, false);
      accIQ[s] = __builtin_amdgcn_wmma_f32_16x16x32_bf16(false, far.v, false, fbi.v,
                                                         (short)0, accIQ[s], false, false);
    }
    __syncthreads();   // all waves done reading buf 'cur' before it is refilled
  }

  // ---- epilogue: C layout = VGPR j: lanes0-15 M=j, lanes16-31 M=j+8, N=lane&15 ----
#pragma unroll
  for (int s = 0; s < 4; ++s) {
    const long col = n0 + wn * 64 + s * 16 + mlow;
#pragma unroll
    for (int j = 0; j < 8; ++j) {
      const long row = m0 + wm * 16 + (half ? 8 : 0) + j;
      const float vr = accRP[s][j] - accRM[s][j];
      const float vi = accIP[s][j] + accIQ[s][j];
      if (OUT_F32) {
        ((float*)Cr_)[row * ldcE + col] = vr;
        ((float*)Ci_)[row * ldcE + col] = vi;
      } else {
        ((bf16_t*)Cr_)[row * ldcE + col] = (bf16_t)vr;
        ((bf16_t*)Ci_)[row * ldcE + col] = (bf16_t)vi;
      }
    }
  }
}

// ---------------------------------------------------------------------------
// Weight transpose + fp32 -> bf16 convert: Wt[n][k] = W[k][n]
// ---------------------------------------------------------------------------
__global__ __launch_bounds__(256) void wconv_kernel(const float* __restrict__ W,
                                                    void* __restrict__ Wt_)
{
  __shared__ float s[32][33];
  bf16_t* Wt = (bf16_t*)Wt_;
  const int tx = threadIdx.x, ty = threadIdx.y;
  const long nb = (long)blockIdx.x * 32;
  const long kb = (long)blockIdx.y * 32;
#pragma unroll
  for (int i = 0; i < 4; ++i)
    s[ty + i * 8][tx] = W[(kb + ty + i * 8) * (long)DIMN + nb + tx];
  __syncthreads();
#pragma unroll
  for (int i = 0; i < 4; ++i)
    Wt[(nb + ty + i * 8) * (long)DIMN + kb + tx] = (bf16_t)s[tx][ty + i * 8];
}

// ---------------------------------------------------------------------------
// ComplexLayerNorm (magnitude norm, phase preserved) + ModReLU, row-wise.
// Reads fp32 rows, writes bf16 IN PLACE into each row's own slot
// (all reads complete before the reduction barrier -> race-free).
// ---------------------------------------------------------------------------
__global__ __launch_bounds__(256) void norm_modrelu_kernel(
    float* __restrict__ zr, float* __restrict__ zi,
    const float* __restrict__ nsc, const float* __restrict__ nsh,
    const float* __restrict__ mrb)
{
  __shared__ float red[256], red2[256];
  const long r = blockIdx.x;
  const int  t = threadIdx.x;
  float* pr = zr + r * (long)DIMN;
  float* pi = zi + r * (long)DIMN;

  float ar[8], ai[8], mg[8];
  float sum = 0.f, sq = 0.f;
#pragma unroll
  for (int i = 0; i < 8; ++i) {
    const int c = t + i * 256;
    const float x = pr[c], y = pi[c];
    ar[i] = x; ai[i] = y;
    const float m = sqrtf(x * x + y * y) + EPSF;
    mg[i] = m; sum += m; sq += m * m;
  }
  red[t] = sum; red2[t] = sq;
  __syncthreads();
  for (int s = 128; s > 0; s >>= 1) {
    if (t < s) { red[t] += red[t + s]; red2[t] += red2[t + s]; }
    __syncthreads();
  }
  const float mean = red[0] * (1.f / DIMN);
  const float var  = (red2[0] - (float)DIMN * mean * mean) * (1.f / (DIMN - 1));
  const float inv  = rsqrtf(var + EPSF);

  bf16_t* qr = (bf16_t*)pr;
  bf16_t* qi = (bf16_t*)pi;
#pragma unroll
  for (int i = 0; i < 8; ++i) {
    const int c = t + i * 256;
    const float nm = (mg[i] - mean) * inv * nsc[c] + nsh[c];
    const float cs = ar[i] / mg[i], sn = ai[i] / mg[i];
    float xr = nm * cs, xi = nm * sn;
    const float nr = sqrtf(xr * xr + xi * xi) + EPSF;
    const float sc = fmaxf(nr + mrb[c], 0.f) / nr;
    qr[c] = (bf16_t)(xr * sc);
    qi[c] = (bf16_t)(xi * sc);
  }
}

// ---------------------------------------------------------------------------
// halt = sigmoid(z_flat @ Wh + bh) ; stack = softmax3(z_flat @ Ws + bs)
// ---------------------------------------------------------------------------
__global__ __launch_bounds__(256) void halt_stack_kernel(
    const float* __restrict__ zr, const float* __restrict__ zi,
    const float* __restrict__ Wh, const float* __restrict__ bh,
    const float* __restrict__ Ws, const float* __restrict__ bs,
    float* __restrict__ halt, float* __restrict__ stack)
{
  __shared__ float4 red[256];
  const long r = blockIdx.x;
  const int  t = threadIdx.x;
  const float* pr = zr + r * (long)DIMN;
  const float* pi = zi + r * (long)DIMN;

  float h = 0.f, s0 = 0.f, s1 = 0.f, s2 = 0.f;
#pragma unroll
  for (int i = 0; i < 8; ++i) {
    const int c = t + i * 256;
    const float v = pr[c];
    h  += v * Wh[c];
    s0 += v * Ws[c * 3 + 0];
    s1 += v * Ws[c * 3 + 1];
    s2 += v * Ws[c * 3 + 2];
  }
#pragma unroll
  for (int i = 0; i < 8; ++i) {
    const int c = t + i * 256;
    const int j = c + DIMN;
    const float v = pi[c];
    h  += v * Wh[j];
    s0 += v * Ws[j * 3 + 0];
    s1 += v * Ws[j * 3 + 1];
    s2 += v * Ws[j * 3 + 2];
  }
  red[t] = make_float4(h, s0, s1, s2);
  __syncthreads();
  for (int s = 128; s > 0; s >>= 1) {
    if (t < s) {
      const float4 a = red[t], b = red[t + s];
      red[t] = make_float4(a.x + b.x, a.y + b.y, a.z + b.z, a.w + b.w);
    }
    __syncthreads();
  }
  if (t == 0) {
    const float4 a = red[0];
    const float hv = 1.f / (1.f + expf(-(a.x + bh[0])));
    const float l0 = a.y + bs[0], l1 = a.z + bs[1], l2 = a.w + bs[2];
    const float m  = fmaxf(l0, fmaxf(l1, l2));
    const float e0 = expf(l0 - m), e1 = expf(l1 - m), e2 = expf(l2 - m);
    const float is = 1.f / (e0 + e1 + e2);
    halt[r] = hv;
    stack[r * 3 + 0] = e0 * is;
    stack[r * 3 + 1] = e1 * is;
    stack[r * 3 + 2] = e2 * is;
  }
}

// ---------------------------------------------------------------------------
extern "C" void kernel_launch(void* const* d_in, const int* in_sizes, int n_in,
                              void* d_out, int out_size, void* d_ws, size_t ws_size,
                              hipStream_t stream)
{
  (void)in_sizes; (void)n_in; (void)out_size; (void)ws_size;

  const float* z_real = (const float*)d_in[0];
  const float* z_imag = (const float*)d_in[1];
  const float* Wlr = (const float*)d_in[2];
  const float* Wli = (const float*)d_in[3];
  const float* nsc = (const float*)d_in[4];
  const float* nsh = (const float*)d_in[5];
  const float* mrb = (const float*)d_in[6];
  // d_in[7..10] = Wq/Wk: mathematically dead (seq_len=1 softmax == 1 exactly)
  const float* Wvr = (const float*)d_in[11];
  const float* Wvi = (const float*)d_in[12];
  const float* Wor = (const float*)d_in[13];
  const float* Woi = (const float*)d_in[14];
  const float* Wh  = (const float*)d_in[15];
  const float* bh  = (const float*)d_in[16];
  const float* Wst = (const float*)d_in[17];
  const float* bs  = (const float*)d_in[18];

  float* out = (float*)d_out;
  const size_t N1 = (size_t)BATCH * DIMN;      // elements per zr/zi tensor
  const size_t WE = (size_t)DIMN * DIMN;       // elements per weight

  // ws: 6 bf16 transposed weights (50.3 MB) + vr/vi bf16 (134.2 MB) = ~185 MB
  char* ws = (char*)d_ws;
  void* WlrT = ws + 0 * WE * 2;
  void* WliT = ws + 1 * WE * 2;
  void* WvrT = ws + 2 * WE * 2;
  void* WviT = ws + 3 * WE * 2;
  void* WorT = ws + 4 * WE * 2;
  void* WoiT = ws + 5 * WE * 2;
  void* bufVr = ws + 6 * WE * 2;
  void* bufVi = ws + 6 * WE * 2 + N1 * 2;

  // 1) weight transpose + bf16 convert (L2-resident afterwards)
  dim3 wgrid(DIMN / 32, DIMN / 32), wblk(32, 8);
  wconv_kernel<<<wgrid, wblk, 0, stream>>>(Wlr, WlrT);
  wconv_kernel<<<wgrid, wblk, 0, stream>>>(Wli, WliT);
  wconv_kernel<<<wgrid, wblk, 0, stream>>>(Wvr, WvrT);
  wconv_kernel<<<wgrid, wblk, 0, stream>>>(Wvi, WviT);
  wconv_kernel<<<wgrid, wblk, 0, stream>>>(Wor, WorT);
  wconv_kernel<<<wgrid, wblk, 0, stream>>>(Woi, WoiT);

  dim3 ggrid(DIMN / TN, BATCH / TM), gblk(256);

  // 2) stage 1: (z_real + i z_imag) @ Wl -> fp32 scratch in d_out
  cgemm_kernel<true, true><<<ggrid, gblk, 0, stream>>>(
      z_real, z_imag, (long)DIMN, WlrT, WliT, out, out + N1, (long)DIMN);

  // 3) ComplexLayerNorm + ModReLU, bf16 written in place (row stride 4096 bf16)
  norm_modrelu_kernel<<<BATCH, 256, 0, stream>>>(out, out + N1, nsc, nsh, mrb);

  // 4) V linear (attention is identity for seq_len=1): bf16 -> bf16 in ws
  cgemm_kernel<false, false><<<ggrid, gblk, 0, stream>>>(
      out, out + N1, (long)(2 * DIMN), WvrT, WviT, bufVr, bufVi, (long)DIMN);

  // 5) output linear: bf16 -> fp32 final zr/zi in d_out
  cgemm_kernel<false, true><<<ggrid, gblk, 0, stream>>>(
      bufVr, bufVi, (long)DIMN, WorT, WoiT, out, out + N1, (long)DIMN);

  // 6) halt + stack heads
  halt_stack_kernel<<<BATCH, 256, 0, stream>>>(
      out, out + N1, Wh, bh, Wst, bs, out + 2 * N1, out + 2 * N1 + BATCH);
}
